// HaloAttention_23072564314758
// MI455X (gfx1250) — compile-verified, matrix operand order checked by reading, not run on hardware
//
#include <hip/hip_runtime.h>
#include <hip/hip_bf16.h>

#define WS_ 16
#define HS_ 8
#define NH_ 4
#define DIM_ 128
#define HD_ 32
#define SCALE_ 0.17677669529663687f  // 1/sqrt(32)

typedef _Float16 h4   __attribute__((ext_vector_type(4)));
typedef _Float16 h8   __attribute__((ext_vector_type(8)));
typedef _Float16 v16h __attribute__((ext_vector_type(16)));
typedef float    f4   __attribute__((ext_vector_type(4)));
typedef float    v8f  __attribute__((ext_vector_type(8)));

__device__ __forceinline__ h8 ld8f_cvt(const float* p) {
  f4 a = *(const f4*)p;
  f4 b = *(const f4*)(p + 4);
  h4 ha = __builtin_convertvector(a, h4);
  h4 hb = __builtin_convertvector(b, h4);
  return __builtin_shufflevector(ha, hb, 0, 1, 2, 3, 4, 5, 6, 7);
}

__device__ __forceinline__ v16h cat16(h8 lo, h8 hi) {
  return __builtin_shufflevector(lo, hi, 0, 1, 2, 3, 4, 5, 6, 7,
                                 8, 9, 10, 11, 12, 13, 14, 15);
}

__device__ __forceinline__ v8f wmma16(v16h a, v16h b, v8f c) {
  return __builtin_amdgcn_wmma_f32_16x16x32_f16(false, a, false, b,
                                                (short)0, c, false, false);
}

// padded coord -> source coord in x: reflect pad (no edge repeat) + roll(-shift)
__device__ __forceinline__ int reflmap(int a, int shift) {
  int r = a - HS_;
  if (r < 0) r = -r;
  if (r > 127) r = 254 - r;
  return (r + shift) & 127;
}

// ---------------------------------------------------------------------------
// Kernel 1: fused window-gather + Q/K/V projection GEMMs (f32 in, f16 out)
// grid (128 windows, 18); one 16-row tile x 128 cols per wave.
// tiles 0..63 -> K (kv rows), 64..127 -> V (kv rows), 128..143 -> Q (center)
// ---------------------------------------------------------------------------
__global__ __launch_bounds__(256) void halo_qkv(
    const float* __restrict__ x,
    const float* __restrict__ Wq, const float* __restrict__ bq,
    const float* __restrict__ Wk, const float* __restrict__ bk,
    const float* __restrict__ Wv, const float* __restrict__ bv,
    const int* __restrict__ shiftp,
    _Float16* __restrict__ Qf, _Float16* __restrict__ Kf,
    _Float16* __restrict__ Vf) {
  const int win = blockIdx.x;
  const int b = win >> 6, wi = (win >> 3) & 7, wj = win & 7;
  const int wave = threadIdx.x >> 5, lane = threadIdx.x & 31;
  const int mrow = lane & 15, hf = lane >> 4;
  const int shift = shiftp[0];
  const int t = blockIdx.y * 8 + wave;  // 0..143

  int rowsel, pi, pj;
  const float *Wm, *bvec;
  _Float16* outp;
  if (t < 64) {
    rowsel = t;
    int row = rowsel * 16 + mrow;
    pi = row >> 5; pj = row & 31;
    Wm = Wk; bvec = bk;
    outp = Kf + ((size_t)win * 1024 + rowsel * 16) * DIM_;
  } else if (t < 128) {
    rowsel = t - 64;
    int row = rowsel * 16 + mrow;
    pi = row >> 5; pj = row & 31;
    Wm = Wv; bvec = bv;
    outp = Vf + ((size_t)win * 1024 + rowsel * 16) * DIM_;
  } else {
    rowsel = t - 128;
    int row = rowsel * 16 + mrow;
    pi = HS_ + (row >> 4); pj = HS_ + (row & 15);
    Wm = Wq; bvec = bq;
    outp = Qf + ((size_t)win * 256 + rowsel * 16) * DIM_;
  }
  const int sr = reflmap(wi * WS_ + pi, shift);
  const int sc = reflmap(wj * WS_ + pj, shift);
  const float* arow = x + (((size_t)b * 128 + sr) * 128 + sc) * DIM_;

  v16h afrag[4];
#pragma unroll
  for (int kt = 0; kt < 4; ++kt) {
    const int kb = kt * 32 + hf * 8;
    afrag[kt] = cat16(ld8f_cvt(arow + kb), ld8f_cvt(arow + kb + 16));
  }
#pragma unroll
  for (int nt = 0; nt < 8; ++nt) {
    const int ncol = nt * 16 + mrow;
    const float bz = bvec[ncol];
    v8f acc;
#pragma unroll
    for (int r = 0; r < 8; ++r) acc[r] = bz;
    const float* wrow = Wm + (size_t)ncol * DIM_;  // B[k][n] = W[n][k]
#pragma unroll
    for (int kt = 0; kt < 4; ++kt) {
      const int kb = kt * 32 + hf * 8;
      v16h bfrag = cat16(ld8f_cvt(wrow + kb), ld8f_cvt(wrow + kb + 16));
      acc = wmma16(afrag[kt], bfrag, acc);
    }
#pragma unroll
    for (int r = 0; r < 8; ++r)
      outp[(size_t)(r + 8 * hf) * DIM_ + ncol] = (_Float16)acc[r];
  }
}

// ---------------------------------------------------------------------------
// Kernel 2: flash-style attention per (window, head). grid 512, 8 waves.
// Each wave: 32 query rows (2 row tiles). 32 chunks of 32 keys.
// ---------------------------------------------------------------------------
__global__ __launch_bounds__(256) void halo_attn(
    const _Float16* __restrict__ Qf, const _Float16* __restrict__ Kf,
    const _Float16* __restrict__ Vf, const float* __restrict__ bias_table,
    _Float16* __restrict__ Obuf) {
  __shared__ __align__(16) _Float16 Vt[32 * 32];   // [d][key] transposed chunk
  __shared__ __align__(16) _Float16 Pb[8][16 * 32];  // per-wave P bounce

  const int win = blockIdx.x >> 2, h = blockIdx.x & 3;
  const int wave = threadIdx.x >> 5, lane = threadIdx.x & 31;
  const int mrow = lane & 15, hf = lane >> 4;
  const int qbase = wave * 32;

  const _Float16* Kwin = Kf + (size_t)win * 1024 * DIM_ + h * HD_;
  const _Float16* Vwin = Vf + (size_t)win * 1024 * DIM_ + h * HD_;

  v16h qfrag[2];
#pragma unroll
  for (int rt = 0; rt < 2; ++rt) {
    const _Float16* qp =
        Qf + ((size_t)win * 256 + qbase + rt * 16 + mrow) * DIM_ + h * HD_;
    qfrag[rt] = cat16(*(const h8*)(qp + hf * 8), *(const h8*)(qp + 16 + hf * 8));
  }

  float mst[2][8], lst[2][8];
  v8f acc[2][2];
#pragma unroll
  for (int rt = 0; rt < 2; ++rt)
#pragma unroll
    for (int r = 0; r < 8; ++r) {
      mst[rt][r] = -1e30f;
      lst[rt][r] = 0.f;
      acc[rt][0][r] = 0.f;
      acc[rt][1][r] = 0.f;
    }

  for (int kc = 0; kc < 32; ++kc) {
    const int base = kc * 32;
    __syncthreads();
#pragma unroll
    for (int u = 0; u < 4; ++u) {  // stage V chunk transposed: Vt[d][kk]
      int idx = u * 256 + threadIdx.x;
      int kk = idx >> 5, d = idx & 31;
      Vt[d * 32 + kk] = Vwin[(size_t)(base + kk) * DIM_ + d];
    }
    __syncthreads();

    v16h kfrag[2];  // B[k=d][n=key]
#pragma unroll
    for (int st = 0; st < 2; ++st) {
      const _Float16* kp = Kwin + (size_t)(base + st * 16 + mrow) * DIM_;
      kfrag[st] =
          cat16(*(const h8*)(kp + hf * 8), *(const h8*)(kp + 16 + hf * 8));
    }

#pragma unroll
    for (int rt = 0; rt < 2; ++rt) {
      float sv[2][8];
#pragma unroll
      for (int st = 0; st < 2; ++st) {
        v8f z;
#pragma unroll
        for (int r = 0; r < 8; ++r) z[r] = 0.f;
        v8f s = wmma16(qfrag[rt], kfrag[st], z);  // one K=32 WMMA = full QK^T
        const int key = base + st * 16 + mrow;    // D layout: col = lane%16
        const int ki = key >> 5, kj = key & 31;
#pragma unroll
        for (int r = 0; r < 8; ++r) {
          const int q = qbase + rt * 16 + r + 8 * hf;  // D row = r + 8*half
          const int pp = 256 + q;                      // ref slice start
          const int d0 = (pp >> 5) - ki + 23;
          const int d1 = (pp & 31) - kj + 23;
          sv[st][r] = s[r] * SCALE_ + bias_table[(d0 * 47 + d1) * NH_ + h];
        }
      }
      // online softmax: rows of this wave-half live in its own 16 lanes
#pragma unroll
      for (int r = 0; r < 8; ++r) {
        float mx = fmaxf(sv[0][r], sv[1][r]);
        mx = fmaxf(mx, __shfl_xor(mx, 1, 16));
        mx = fmaxf(mx, __shfl_xor(mx, 2, 16));
        mx = fmaxf(mx, __shfl_xor(mx, 4, 16));
        mx = fmaxf(mx, __shfl_xor(mx, 8, 16));
        const float mnew = fmaxf(mst[rt][r], mx);
        const float alpha = __expf(mst[rt][r] - mnew);
        const float p0 = __expf(sv[0][r] - mnew);
        const float p1 = __expf(sv[1][r] - mnew);
        float ls = p0 + p1;
        ls += __shfl_xor(ls, 1, 16);
        ls += __shfl_xor(ls, 2, 16);
        ls += __shfl_xor(ls, 4, 16);
        ls += __shfl_xor(ls, 8, 16);
        mst[rt][r] = mnew;
        lst[rt][r] = lst[rt][r] * alpha + ls;
        acc[rt][0][r] *= alpha;
        acc[rt][1][r] *= alpha;
        sv[0][r] = p0;
        sv[1][r] = p1;
      }
      // D-layout -> A-layout transpose of P via per-wave LDS bounce
      _Float16* pb = &Pb[wave][0];
#pragma unroll
      for (int st = 0; st < 2; ++st)
#pragma unroll
        for (int r = 0; r < 8; ++r)
          pb[(r + 8 * hf) * 32 + st * 16 + mrow] = (_Float16)sv[st][r];
      asm volatile("s_wait_dscnt 0" ::: "memory");  // wave-local LDS turnaround
      v16h pfrag = cat16(*(const h8*)(pb + mrow * 32 + hf * 8),
                         *(const h8*)(pb + mrow * 32 + 16 + hf * 8));
#pragma unroll
      for (int ct = 0; ct < 2; ++ct) {  // B[k=key][n=d] from transposed Vt
        const _Float16* vp = &Vt[(ct * 16 + mrow) * 32];
        v16h vfrag =
            cat16(*(const h8*)(vp + hf * 8), *(const h8*)(vp + 16 + hf * 8));
        acc[rt][ct] = wmma16(pfrag, vfrag, acc[rt][ct]);
      }
    }
  }

#pragma unroll
  for (int rt = 0; rt < 2; ++rt)
#pragma unroll
    for (int ct = 0; ct < 2; ++ct)
#pragma unroll
      for (int r = 0; r < 8; ++r) {
        const int q = qbase + rt * 16 + r + 8 * hf;
        const int d = ct * 16 + mrow;
        Obuf[((size_t)win * 256 + q) * DIM_ + h * HD_ + d] =
            (_Float16)(acc[rt][ct][r] / lst[rt][r]);
      }
}

// ---------------------------------------------------------------------------
// Kernel 3: output projection + window->(B,H,W,C) scatter (with roll(+shift))
// grid (128, 2); one 16-row tile per wave.
// ---------------------------------------------------------------------------
__global__ __launch_bounds__(256) void halo_proj(
    const _Float16* __restrict__ Obuf, const float* __restrict__ Wp,
    const float* __restrict__ bp, const int* __restrict__ shiftp,
    float* __restrict__ out) {
  const int win = blockIdx.x;
  const int b = win >> 6, wi = (win >> 3) & 7, wj = win & 7;
  const int wave = threadIdx.x >> 5, lane = threadIdx.x & 31;
  const int mrow = lane & 15, hf = lane >> 4;
  const int shift = shiftp[0];
  const int rowbase = (blockIdx.y * 8 + wave) * 16;  // 0..240

  const _Float16* arow = Obuf + ((size_t)win * 256 + rowbase + mrow) * DIM_;
  v16h afrag[4];
#pragma unroll
  for (int kt = 0; kt < 4; ++kt) {
    const int kb = kt * 32 + hf * 8;
    afrag[kt] = cat16(*(const h8*)(arow + kb), *(const h8*)(arow + kb + 16));
  }
#pragma unroll
  for (int nt = 0; nt < 8; ++nt) {
    const int ncol = nt * 16 + mrow;
    const float bz = bp[ncol];
    v8f acc;
#pragma unroll
    for (int r = 0; r < 8; ++r) acc[r] = bz;
    const float* wrow = Wp + (size_t)ncol * DIM_;
#pragma unroll
    for (int kt = 0; kt < 4; ++kt) {
      const int kb = kt * 32 + hf * 8;
      v16h bfrag = cat16(ld8f_cvt(wrow + kb), ld8f_cvt(wrow + kb + 16));
      acc = wmma16(afrag[kt], bfrag, acc);
    }
#pragma unroll
    for (int r = 0; r < 8; ++r) {
      const int q = rowbase + r + 8 * hf;
      const int hh = (wi * WS_ + (q >> 4) + shift) & 127;
      const int ww = (wj * WS_ + (q & 15) + shift) & 127;
      out[(((size_t)b * 128 + hh) * 128 + ww) * DIM_ + ncol] = acc[r];
    }
  }
}

extern "C" void kernel_launch(void* const* d_in, const int* in_sizes, int n_in,
                              void* d_out, int out_size, void* d_ws,
                              size_t ws_size, hipStream_t stream) {
  const float* x = (const float*)d_in[0];
  const float* Wq = (const float*)d_in[1];
  const float* bq = (const float*)d_in[2];
  const float* Wk = (const float*)d_in[3];
  const float* bk = (const float*)d_in[4];
  const float* Wv = (const float*)d_in[5];
  const float* bv = (const float*)d_in[6];
  const float* Wp = (const float*)d_in[7];
  const float* bp = (const float*)d_in[8];
  const float* bt = (const float*)d_in[9];
  const int* shift = (const int*)d_in[10];

  char* ws = (char*)d_ws;
  const size_t QB = (size_t)128 * 256 * 128 * sizeof(_Float16);   // 8 MiB
  const size_t KB = (size_t)128 * 1024 * 128 * sizeof(_Float16);  // 32 MiB
  _Float16* Qf = (_Float16*)(ws);
  _Float16* Kf = (_Float16*)(ws + QB);
  _Float16* Vf = (_Float16*)(ws + QB + KB);
  _Float16* Ob = (_Float16*)(ws + QB + 2 * KB);  // total 80 MiB

  halo_qkv<<<dim3(128, 18), 256, 0, stream>>>(x, Wq, bq, Wk, bk, Wv, bv, shift,
                                              Qf, Kf, Vf);
  halo_attn<<<dim3(512), 256, 0, stream>>>(Qf, Kf, Vf, bt, Ob);
  halo_proj<<<dim3(128, 2), 256, 0, stream>>>(Ob, Wp, bp, shift,
                                              (float*)d_out);
}
